// AutoencoderVQ_87247965651414
// MI455X (gfx1250) — matrix operand order repaired and checked by reading, not compiled, not run on hardware
//
#include <hip/hip_runtime.h>

// ---------------- types ----------------
typedef __attribute__((ext_vector_type(8)))  __bf16 v8bf;
typedef __attribute__((ext_vector_type(16))) __bf16 v16bf;
typedef __attribute__((ext_vector_type(8)))  float  v8f;

#define NUM_CODES  8192
#define LATENT     256
#define NROWS      16384          // 16*32*32
#define ROWS_PER_BLK 128
#define KSPLIT     8
#define KCHUNK     16             // codes per k-iteration (one WMMA N-tile)
#define LDSTRIDE   264            // 256 + 8 pad: conflict-free b128 LDS reads
#define ABUF (ROWS_PER_BLK * LDSTRIDE)   // 33792 elems: one A plane
#define BBUF (KCHUNK * LDSTRIDE)         //  4224 elems: one B plane

// ---------------------------------------------------------------------------
// Kernel 1: split codebook fp32 -> bf16 hi/lo, exact fp32 row norms ||e||^2
// ---------------------------------------------------------------------------
__global__ void vq_prep_codebook(const float* __restrict__ cb,
                                 __bf16* __restrict__ Eh,
                                 __bf16* __restrict__ El,
                                 float*  __restrict__ Enorm) {
    __shared__ float red[256];
    const int row = blockIdx.x;
    const int t   = threadIdx.x;
    const float x = cb[(size_t)row * LATENT + t];
    const __bf16 h = (__bf16)x;                 // RNE truncation
    const __bf16 l = (__bf16)(x - (float)h);    // residual
    Eh[(size_t)row * LATENT + t] = h;
    El[(size_t)row * LATENT + t] = l;
    red[t] = x * x;
    __syncthreads();
    for (int s = 128; s > 0; s >>= 1) {
        if (t < s) red[t] += red[t + s];
        __syncthreads();
    }
    if (t == 0) Enorm[row] = red[0];
}

// B-tile staging into LDS at integer offset bufOff (keeps addrspace(3) loads):
// 16 codes x 256 C (hi at bufOff, lo at bufOff+BBUF), coalesced b128 + prefetch
__device__ __forceinline__ void stage_b(__bf16* __restrict__ smem, int bufOff,
                                        const __bf16* __restrict__ Eh,
                                        const __bf16* __restrict__ El,
                                        int k0, int kend, int tid) {
    const int code_t = tid >> 4;          // 0..15
    const int c0     = (tid & 15) * 16;   // 0..240
    const int soff   = bufOff + code_t * LDSTRIDE + c0;
    const uint4* gh = (const uint4*)(Eh + (size_t)(k0 + code_t) * LATENT + c0);
    uint4* sh = (uint4*)(smem + soff);
    sh[0] = gh[0]; sh[1] = gh[1];
    const uint4* gl = (const uint4*)(El + (size_t)(k0 + code_t) * LATENT + c0);
    uint4* sl = (uint4*)(smem + soff + BBUF);
    sl[0] = gl[0]; sl[1] = gl[1];
    if (k0 + KCHUNK < kend) {             // -> global_prefetch_b8
        __builtin_prefetch(Eh + (size_t)(k0 + KCHUNK + code_t) * LATENT + c0, 0, 1);
        __builtin_prefetch(El + (size_t)(k0 + KCHUNK + code_t) * LATENT + c0, 0, 1);
    }
}

// ---------------------------------------------------------------------------
// Kernel 2: fused GEMM + running argmin, split-bf16 WMMA.
// A fragments (hi+lo) hoisted to registers for the whole k-stream; B tiles
// double-buffered in LDS (one barrier/iter, staging overlaps WMMA). The A
// staging region is reused as the B double buffer after the hoist.
// ---------------------------------------------------------------------------
__global__ void __launch_bounds__(256, 1)
vq_argmin_gemm(const float* __restrict__ z,
               const __bf16* __restrict__ Eh,
               const __bf16* __restrict__ El,
               const float*  __restrict__ Enorm,
               float* __restrict__ pdist,
               int*   __restrict__ pidx) {
    extern __shared__ char smem_raw[];
    __bf16* smem = (__bf16*)smem_raw;
    // phase 1: As_h at [0, ABUF), As_l at [ABUF, 2*ABUF)
    // phase 2: B double buffer aliases [0, 4*BBUF): buf i -> hi at i*2*BBUF, lo +BBUF

    const int tid      = threadIdx.x;
    const int lane     = tid & 31;
    const int wid      = tid >> 5;          // 0..7 : wave's 16-row tile
    const int laneHalf = lane >> 4;         // 0/1
    const int laneMod  = lane & 15;

    const int blk_n = blockIdx.x & 127;     // row-block
    const int part  = blockIdx.x >> 7;      // k partition 0..KSPLIT-1
    const int n0    = blk_n * ROWS_PER_BLK;
    const int b     = n0 >> 10;             // batch index (128 | 1024)
    const int hw0   = n0 & 1023;
    const int kbeg  = part * (NUM_CODES / KSPLIT);
    const int kend  = kbeg + (NUM_CODES / KSPLIT);
    const int nIter = (kend - kbeg) / KCHUNK;

    // ---- phase 1: stage A (transpose (b,c,hw) -> [row][c], split bf16) ----
    {
        const int hw_off = (tid & 31) * 4;   // 4 consecutive hw per thread
        const int cq     = tid >> 5;         // 0..7
        const float* zb  = z + (size_t)b * (LATENT * 1024) + hw0;
        for (int co = 0; co < LATENT; co += 8) {
            const int c = co + cq;
            const float4 f = *(const float4*)(zb + (size_t)c * 1024 + hw_off);
            const float xs[4] = {f.x, f.y, f.z, f.w};
#pragma unroll
            for (int j = 0; j < 4; ++j) {
                const float  x = xs[j];
                const __bf16 h = (__bf16)x;
                const __bf16 l = (__bf16)(x - (float)h);
                smem[(hw_off + j) * LDSTRIDE + c]        = h;   // As_h
                smem[ABUF + (hw_off + j) * LDSTRIDE + c] = l;   // As_l
            }
        }
    }
    __syncthreads();

    // ---- hoist A fragments into registers (loop-invariant over k) ----
    // A layout (16x32 bf16): lane<16 holds K{0..7,16..23}, lane>=16 K{8..15,24..31}
    const int arow = wid * KCHUNK + laneMod;   // A row in LDS (0..127)
    v16bf Ah[8], Al[8];
#pragma unroll
    for (int ci = 0; ci < 8; ++ci) {
        const int aoff = arow * LDSTRIDE + ci * 32 + laneHalf * 8;
        const v8bf ahl = *(const v8bf*)(smem + aoff);
        const v8bf ahh = *(const v8bf*)(smem + aoff + 16);
        const v8bf all = *(const v8bf*)(smem + ABUF + aoff);
        const v8bf alh = *(const v8bf*)(smem + ABUF + aoff + 16);
        Ah[ci] = __builtin_shufflevector(ahl, ahh, 0,1,2,3,4,5,6,7,8,9,10,11,12,13,14,15);
        Al[ci] = __builtin_shufflevector(all, alh, 0,1,2,3,4,5,6,7,8,9,10,11,12,13,14,15);
    }
    __syncthreads();   // all waves done reading A before B staging overwrites LDS

    // running argmin state: 8 accumulator rows per lane
    float best_s[8];
    int   best_i[8];
#pragma unroll
    for (int r = 0; r < 8; ++r) { best_s[r] = 3.4e38f; best_i[r] = 0; }

    const int bcol = laneMod;                  // code column within tile

    // prologue: stage first B tile into buffer 0
    stage_b(smem, 0, Eh, El, kbeg, kend, tid);

    // ---- stream over code tiles (double-buffered, one barrier per iter) ----
    for (int it = 0; it < nIter; ++it) {
        const int k0      = kbeg + it * KCHUNK;
        const int bufBase = (it & 1) * (2 * BBUF);
        __syncthreads();   // stage(it) visible; compute(it-1) reads done
        if (it + 1 < nIter)
            stage_b(smem, bufBase ^ (2 * BBUF), Eh, El, k0 + KCHUNK, kend, tid);

        v8f acc = {};   // fp32 16x16 dot tile
#pragma unroll
        for (int ci = 0; ci < 8; ++ci) {
            // B fragment (32x16 bf16): lane<16 -> K{0..7,16..23} of its column, lane>=16 -> K{8..15,24..31}
            const int boff = bufBase + bcol * LDSTRIDE + ci * 32 + laneHalf * 8;
            const v8bf bhl = *(const v8bf*)(smem + boff);
            const v8bf bhh = *(const v8bf*)(smem + boff + 16);
            const v8bf bll = *(const v8bf*)(smem + boff + BBUF);
            const v8bf blh = *(const v8bf*)(smem + boff + BBUF + 16);
            const v16bf b_h = __builtin_shufflevector(bhl, bhh, 0,1,2,3,4,5,6,7,8,9,10,11,12,13,14,15);
            const v16bf b_l = __builtin_shufflevector(bll, blh, 0,1,2,3,4,5,6,7,8,9,10,11,12,13,14,15);
            // split-bf16 fp32 emulation: hh + hl + lh
            acc = __builtin_amdgcn_wmma_f32_16x16x32_bf16(false, Ah[ci], false, b_h, (short)0, acc, false, false);
            acc = __builtin_amdgcn_wmma_f32_16x16x32_bf16(false, Ah[ci], false, b_l, (short)0, acc, false, false);
            acc = __builtin_amdgcn_wmma_f32_16x16x32_bf16(false, Al[ci], false, b_h, (short)0, acc, false, false);
        }

        // epilogue: score = ||e||^2 - 2*dot ; running argmin (strict < keeps lowest idx)
        const int   code = k0 + bcol;
        const float nrm  = Enorm[code];
#pragma unroll
        for (int r = 0; r < 8; ++r) {
            const float s = fmaf(-2.0f, acc[r], nrm);
            if (s < best_s[r]) { best_s[r] = s; best_i[r] = code; }
        }
    }

    // ---- reduce across the 16 code-columns (lanes within each half) ----
#pragma unroll
    for (int r = 0; r < 8; ++r) {
        float s = best_s[r];
        int   i = best_i[r];
#pragma unroll
        for (int m = 8; m >= 1; m >>= 1) {
            const float s2 = __shfl_xor(s, m, 32);
            const int   i2 = __shfl_xor(i, m, 32);
            if (s2 < s || (s2 == s && i2 < i)) { s = s2; i = i2; }
        }
        best_s[r] = s;
        best_i[r] = i;
    }
    if (laneMod == 0) {
        const int base = n0 + wid * KCHUNK + laneHalf * 8;   // D row = r + 8*half
#pragma unroll
        for (int r = 0; r < 8; ++r) {
            pdist[(size_t)(base + r) * KSPLIT + part] = best_s[r];
            pidx [(size_t)(base + r) * KSPLIT + part] = best_i[r];
        }
    }
}

// ---------------------------------------------------------------------------
// Kernel 3: combine k-partials, gather code rows, write (B,C,H,W) output
// with straight-through rounding, accumulate 1.25*mean(diff^2).
// ---------------------------------------------------------------------------
__global__ void vq_output(const float* __restrict__ z,
                          const float* __restrict__ cbf,
                          const float* __restrict__ pdist,
                          const int*   __restrict__ pidx,
                          float* __restrict__ out,
                          float* __restrict__ loss) {
    __shared__ float red[256];
    const int t  = threadIdx.x;
    const int b  = blockIdx.x >> 2;
    const int hw = (blockIdx.x & 3) * 256 + t;
    const int n  = b * 1024 + hw;

    float bd = 3.4e38f;
    int   id = 0;
#pragma unroll
    for (int p = 0; p < KSPLIT; ++p) {      // ascending p => lowest index on ties
        const float d = pdist[(size_t)n * KSPLIT + p];
        const int   i = pidx [(size_t)n * KSPLIT + p];
        if (d < bd) { bd = d; id = i; }
    }

    const float* crow = cbf + (size_t)id * LATENT;
    const float* zb   = z   + (size_t)b * (LATENT * 1024) + hw;
    float*       ob   = out + (size_t)b * (LATENT * 1024) + hw;
    float acc = 0.f;
    for (int c = 0; c < LATENT; ++c) {
        const float cv = crow[c];
        const float zv = zb[(size_t)c * 1024];
        const float d  = cv - zv;
        acc = fmaf(d, d, acc);
        ob[(size_t)c * 1024] = zv + (cv - zv);   // straight-through rounding
    }

    red[t] = acc;
    __syncthreads();
    for (int s = 128; s > 0; s >>= 1) {
        if (t < s) red[t] += red[t + s];
        __syncthreads();
    }
    if (t == 0) atomicAdd(loss, red[0] * (1.25f / 4194304.f));
}

// ---------------------------------------------------------------------------
extern "C" void kernel_launch(void* const* d_in, const int* in_sizes, int n_in,
                              void* d_out, int out_size, void* d_ws, size_t ws_size,
                              hipStream_t stream) {
    (void)in_sizes; (void)n_in; (void)out_size; (void)ws_size;
    const float* z  = (const float*)d_in[0];   // (16,256,32,32)
    const float* cb = (const float*)d_in[1];   // (8192,256)
    float* out = (float*)d_out;                // 4194304 z_q + 1 loss

    char* ws = (char*)d_ws;
    __bf16* Eh    = (__bf16*)(ws);
    __bf16* El    = (__bf16*)(ws + 4194304);
    float*  Enorm = (float*) (ws + 8388608);
    float*  pdist = (float*) (ws + 8388608 + 32768);
    int*    pidx  = (int*)   (ws + 8388608 + 32768 + (size_t)NROWS * KSPLIT * 4);

    vq_prep_codebook<<<NUM_CODES, 256, 0, stream>>>(cb, Eh, El, Enorm);

    const size_t smem = (size_t)(2 * ABUF) * sizeof(__bf16); // 135168 B
    vq_argmin_gemm<<<(NROWS / ROWS_PER_BLK) * KSPLIT, 256, smem, stream>>>(
        z, Eh, El, Enorm, pdist, pidx);

    hipMemsetAsync(out + 4194304, 0, sizeof(float), stream);
    vq_output<<<64, 256, 0, stream>>>(z, cb, pdist, pidx, out, out + 4194304);
}
